// Attention_Layer_87806311399815
// MI455X (gfx1250) — compile-verified, matrix operand order checked by reading, not run on hardware
//
#include <hip/hip_runtime.h>

// ---------- types ----------
typedef __attribute__((ext_vector_type(16))) __bf16 v16bf;
typedef __attribute__((ext_vector_type(8)))  __bf16 v8bf;
typedef __attribute__((ext_vector_type(8)))  float  v8f;

#define HDIM   256
#define PDIM   128
#define NHEADS 4
#define NPTS   16384
#define NPB    2048   // points per frame
#define DH     64

__device__ __forceinline__ v8f wmma_bf16(v16bf a, v16bf b, v8f c) {
    return __builtin_amdgcn_wmma_f32_16x16x32_bf16(false, a, false, b, (short)0, c, false, false);
}

// Load one 16x32 bf16 fragment slice for this lane per the CDNA5 16-bit A layout:
// lanes 0-15 hold K = {0..7, 16..23}; lanes 16-31 hold K = {8..15, 24..31}.
// Caller folds the (lane>>4)*8 K-offset into p; chunks are at p[0..7] and p[16..23].
__device__ __forceinline__ v16bf ldfrag(const __bf16* p) {
    union { v16bf v; v8bf h[2]; } u;
    u.h[0] = *(const v8bf*)(p);
    u.h[1] = *(const v8bf*)(p + 16);
    return u.v;
}
__device__ __forceinline__ v16bf ldfrag_lds(const unsigned short* p) {
    union { v16bf v; v8bf h[2]; } u;
    u.h[0] = *(const v8bf*)(p);
    u.h[1] = *(const v8bf*)(p + 16);
    return u.v;
}
__device__ __forceinline__ unsigned short f2bf_bits(float f) {
    union { __bf16 b; unsigned short u; } cv; cv.b = (__bf16)f; return cv.u;
}

// Max-reduce across each 16-lane half using DPP16 only (no LDS, no waits):
// quad_perm xor1, quad_perm xor2, row_half_mirror, row_mirror.
__device__ __forceinline__ float halfmax_dpp(float x) {
    int t;
    t = __builtin_amdgcn_update_dpp(0, __float_as_int(x), 0xB1,  0xF, 0xF, true); // quad_perm [1,0,3,2]
    x = fmaxf(x, __int_as_float(t));
    t = __builtin_amdgcn_update_dpp(0, __float_as_int(x), 0x4E,  0xF, 0xF, true); // quad_perm [2,3,0,1]
    x = fmaxf(x, __int_as_float(t));
    t = __builtin_amdgcn_update_dpp(0, __float_as_int(x), 0x141, 0xF, 0xF, true); // row_half_mirror
    x = fmaxf(x, __int_as_float(t));
    t = __builtin_amdgcn_update_dpp(0, __float_as_int(x), 0x140, 0xF, 0xF, true); // row_mirror
    x = fmaxf(x, __int_as_float(t));
    return x;
}

// ---------- elementwise converts ----------
__global__ void cvt_bf16(const float* __restrict__ s, __bf16* __restrict__ d, int n) {
    int i = blockIdx.x * blockDim.x + threadIdx.x;
    if (i < n) d[i] = (__bf16)s[i];
}
// copy f32 [rows,cols] into bf16 [rows,ldd] (column offset applied by caller via d)
__global__ void cvt_pad(const float* __restrict__ s, __bf16* __restrict__ d,
                        int rows, int cols, int ldd) {
    int i = blockIdx.x * blockDim.x + threadIdx.x;
    if (i >= rows * cols) return;
    int r = i / cols, c = i - r * cols;
    d[(size_t)r * ldd + c] = (__bf16)s[i];
}

// ---------- positional embedding: pos2embed ----------
// e[i][2j]     = sin(2*pi*y / t_j),  e[i][2j+1]     = cos(2*pi*y / t_j)
// e[i][128+2j] = sin(2*pi*x / t_j),  e[i][128+2j+1] = cos(2*pi*x / t_j)
// t_j = 1 + j/64,  y = coords[i][2], x = coords[i][3]
__global__ void embed_kernel(const float* __restrict__ coords, __bf16* __restrict__ e, int npts) {
    int idx = blockIdx.x * blockDim.x + threadIdx.x;
    if (idx >= npts * 64) return;
    int i = idx >> 6, j = idx & 63;
    float y = coords[i * 4 + 2] * 6.28318530718f;
    float x = coords[i * 4 + 3] * 6.28318530718f;
    float t = 1.0f + (float)j * (1.0f / 64.0f);
    float rt = __frcp_rn(t);
    float ay = y * rt, ax = x * rt;
    __bf16* row = e + (size_t)i * 256;
    row[2 * j]           = (__bf16)__sinf(ay);
    row[2 * j + 1]       = (__bf16)__cosf(ay);
    row[128 + 2 * j]     = (__bf16)__sinf(ax);
    row[128 + 2 * j + 1] = (__bf16)__cosf(ax);
}

// ---------- generic WMMA GEMM: C[M,N] = act((A[M,K] @ W[N,K]^T + bias) * oscale) ----------
// block = 128 threads (4 waves). wave tile = 16 (M) x 64 (N) = 4 C-tiles.
enum { EPI_BF16 = 0, EPI_RELU_BF16 = 1, EPI_F32 = 2, EPI_VT_SCATTER = 3 };

template <int EPI>
__global__ __launch_bounds__(128) void gemm_wmma(
    const __bf16* __restrict__ A, const __bf16* __restrict__ W,
    const float* __restrict__ bias, void* __restrict__ Cout,
    int M, int N, int K, int ldc, float oscale)
{
    const int lane  = threadIdx.x & 31;
    const int wv    = threadIdx.x >> 5;
    const int mbase = blockIdx.x * 16;
    const int nbase = blockIdx.y * 256 + wv * 64;
    if (nbase >= N) return;                 // wave-uniform

    const int koff = (lane >> 4) * 8;
    const __bf16* arow = A + (size_t)(mbase + (lane & 15)) * K + koff;
    const __bf16* wrow = W + (size_t)(nbase + (lane & 15)) * K + koff;

    v8f c[4] = {v8f{}, v8f{}, v8f{}, v8f{}};
    for (int k0 = 0; k0 < K; k0 += 32) {
        v16bf a = ldfrag(arow + k0);
#pragma unroll
        for (int t = 0; t < 4; ++t) {
            v16bf b = ldfrag(wrow + (size_t)(t * 16) * K + k0);
            c[t] = wmma_bf16(a, b, c[t]);
        }
    }

    float*  co32 = (float*)Cout;
    __bf16* co16 = (__bf16*)Cout;
    const int rbase = mbase + ((lane >> 4) * 8);   // C layout: lanes 16-31 -> M += 8
#pragma unroll
    for (int t = 0; t < 4; ++t) {
        int col = nbase + t * 16 + (lane & 15);
        float bb = bias ? bias[col] : 0.0f;
#pragma unroll
        for (int r = 0; r < 8; ++r) {
            int row = rbase + r;
            float val = (c[t][r] + bb) * oscale;
            if (EPI == EPI_RELU_BF16) val = fmaxf(val, 0.0f);
            if (EPI == EPI_F32) {
                co32[(size_t)row * ldc + col] = val;
            } else if (EPI == EPI_VT_SCATTER) {
                // v2[row, col] -> vT[b, head, d, kr] with kr = row%2048, b = row/2048
                int b = row >> 11, kr = row & 2047, h = col >> 6, d = col & 63;
                co16[(((size_t)(b * NHEADS + h) * DH + d) * NPB) + kr] = (__bf16)val;
            } else {
                co16[(size_t)row * ldc + col] = (__bf16)val;
            }
        }
    }
}

// ---------- flash attention ----------
// q2 (pre-scaled by 1/sqrt(dh)), k2: [B*npb, 256] row-major (head h = cols h*64..),
// vT: [B,H,64,2048], o: [B*npb,256] bf16.
// Online softmax with a per-16-lane-half running max (each query row's 16 score
// columns live entirely within one half, so a per-half max is a valid per-row
// upper bound). Row denominators accumulated by the tensor core via a constant
// ones-column operand.
__global__ __launch_bounds__(128) void flash_attn(
    const __bf16* __restrict__ q2, const __bf16* __restrict__ k2,
    const __bf16* __restrict__ vT, __bf16* __restrict__ o)
{
    __shared__ alignas(16) unsigned char smem[4 * 16 * 80];   // per-wave 16x32 bf16 P tile, 80B row stride
    const int lane = threadIdx.x & 31;
    const int wv   = threadIdx.x >> 5;
    const int bh = blockIdx.x, b = bh >> 2, h = bh & 3;
    const int qbase = blockIdx.y * 64 + wv * 16;
    const int koff = (lane >> 4) * 8;
    const int rsel = (lane >> 4) * 8;          // +8 row offset for lanes 16-31 (C layout)

    // Q A-fragments: d = 0..31 and 32..63
    const __bf16* qrow = q2 + (size_t)(b * NPB + qbase + (lane & 15)) * HDIM + h * DH + koff;
    v16bf aq0 = ldfrag(qrow);
    v16bf aq1 = ldfrag(qrow + 32);

    const __bf16* kbase = k2 + (size_t)(b * NPB) * HDIM + h * DH + koff;
    const __bf16* vbase = vT + (size_t)(b * NHEADS + h) * DH * NPB + koff;
    unsigned short* pl = (unsigned short*)smem + wv * (16 * 40);

    // constant B-fragment: column 0 = ones, columns 1..15 = zeros
    // (B lanes 0/16 own column 0 -> set their 16 K-values to 1.0)
    v16bf onesf = {};
    {
        __bf16 e = (__bf16)(((lane & 15) == 0) ? 1.0f : 0.0f);
#pragma unroll
        for (int i = 0; i < 16; ++i) onesf[i] = e;
    }

    float m = -1e30f;
    v8f acc[5] = {v8f{}, v8f{}, v8f{}, v8f{}, v8f{}};   // [0..3]=O d-tiles, [4]=rowsum

    for (int kt = 0; kt < NPB; kt += 32) {
        // S = q . k^T (scale already folded into q2), two 16x16 tiles
        const __bf16* k0p = kbase + (size_t)(kt + (lane & 15)) * HDIM;
        const __bf16* k1p = k0p + (size_t)16 * HDIM;
        v8f s0{}, s1{};
        s0 = wmma_bf16(aq0, ldfrag(k0p), s0);
        s0 = wmma_bf16(aq1, ldfrag(k0p + 32), s0);
        s1 = wmma_bf16(aq0, ldfrag(k1p), s1);
        s1 = wmma_bf16(aq1, ldfrag(k1p + 32), s1);

        // per-half running max: local 16-value tree + 4 DPP16 steps (no LDS)
        float mx = fmaxf(s0[0], s1[0]);
#pragma unroll
        for (int r = 1; r < 8; ++r) mx = fmaxf(mx, fmaxf(s0[r], s1[r]));
        mx = halfmax_dpp(mx);
        float mnew = fmaxf(m, mx);
        float scale = __expf(m - mnew);
        m = mnew;
#pragma unroll
        for (int t = 0; t < 5; ++t) {
#pragma unroll
            for (int r = 0; r < 8; ++r) acc[t][r] *= scale;
        }

        // P = exp(S - m), staged row-major in LDS (C-layout -> A-layout re-shuffle)
#pragma unroll
        for (int r = 0; r < 8; ++r) {
            float p0 = __expf(s0[r] - mnew);
            float p1 = __expf(s1[r] - mnew);
            unsigned short* prw = pl + (size_t)(r + rsel) * 40 + (lane & 15);
            prw[0]  = f2bf_bits(p0);
            prw[16] = f2bf_bits(p1);
        }

        // reload P as A-fragment (K = 32 keys)
        v16bf pa = ldfrag_lds(pl + (size_t)(lane & 15) * 40 + koff);

        // O += P @ V ; rowsum += P @ ones
#pragma unroll
        for (int t = 0; t < 4; ++t) {
            const __bf16* vp = vbase + (size_t)(t * 16 + (lane & 15)) * NPB + kt;
            acc[t] = wmma_bf16(pa, ldfrag(vp), acc[t]);
        }
        acc[4] = wmma_bf16(pa, onesf, acc[4]);
    }

    // row denominators live in column 0 of acc[4]: lane 0 (rows 0-7) / lane 16 (rows 8-15)
    float invl[8];
#pragma unroll
    for (int r = 0; r < 8; ++r) {
        float l = __shfl(acc[4][r], lane & 16);
        invl[r] = __frcp_rn(l);
    }
#pragma unroll
    for (int t = 0; t < 4; ++t) {
#pragma unroll
        for (int r = 0; r < 8; ++r) {
            int row = qbase + r + rsel;
            float val = acc[t][r] * invl[r];
            o[(size_t)(b * NPB + row) * HDIM + h * DH + t * 16 + (lane & 15)] = (__bf16)val;
        }
    }
}

// ---------- LayerNorm over concat(x_q, o_proj), width 512, one wave per row ----------
__global__ __launch_bounds__(128) void ln_kernel(
    const float* __restrict__ xq, const float* __restrict__ oproj,
    const float* __restrict__ w, const float* __restrict__ bln,
    __bf16* __restrict__ feat)
{
    int row  = blockIdx.x * (blockDim.x >> 5) + (threadIdx.x >> 5);
    int lane = threadIdx.x & 31;
    const float* xr   = xq    + (size_t)row * 256;
    const float* orow = oproj + (size_t)row * 256;
    float v[16], s = 0.0f, s2 = 0.0f;
    int base = lane * 16;
#pragma unroll
    for (int e = 0; e < 16; ++e) {
        int c = base + e;
        float val = (c < 256) ? xr[c] : orow[c - 256];
        v[e] = val; s += val; s2 += val * val;
    }
#pragma unroll
    for (int off = 1; off < 32; off <<= 1) { s += __shfl_xor(s, off); s2 += __shfl_xor(s2, off); }
    float mean = s * (1.0f / 512.0f);
    float var  = s2 * (1.0f / 512.0f) - mean * mean;
    float inv  = rsqrtf(var + 1e-5f);
#pragma unroll
    for (int e = 0; e < 16; ++e) {
        int c = base + e;
        float nv = (v[e] - mean) * inv * w[c] + bln[c];
        feat[(size_t)row * 512 + c] = (__bf16)nv;
    }
}

// ---------- host ----------
static inline dim3 gemm_grid(int M, int N) { return dim3(M / 16, (N + 255) / 256); }

extern "C" void kernel_launch(void* const* d_in, const int* in_sizes, int n_in,
                              void* d_out, int out_size, void* d_ws, size_t ws_size,
                              hipStream_t stream) {
    (void)in_sizes; (void)n_in; (void)out_size; (void)ws_size;
    const float* inputs  = (const float*)d_in[0];
    const float* Q_in    = (const float*)d_in[1];
    const float* icoords = (const float*)d_in[2];
    const float* qcoords = (const float*)d_in[3];
    const float* Wq   = (const float*)d_in[4];
    const float* Wk   = (const float*)d_in[5];
    const float* Wv   = (const float*)d_in[6];
    const float* peW1 = (const float*)d_in[7];
    const float* peb1 = (const float*)d_in[8];
    const float* peW2 = (const float*)d_in[9];
    const float* peb2 = (const float*)d_in[10];
    const float* inW  = (const float*)d_in[11];
    const float* inb  = (const float*)d_in[12];
    const float* outW = (const float*)d_in[13];
    const float* outb = (const float*)d_in[14];
    const float* lnw  = (const float*)d_in[15];
    const float* lnb  = (const float*)d_in[16];
    const float* alnW = (const float*)d_in[17];

    char* base = (char*)d_ws;
    size_t cur = 0;
    auto alloc = [&](size_t bytes) -> void* {
        void* p = base + cur;
        cur += (bytes + 255) & ~(size_t)255;
        return p;
    };
    __bf16* bWq   = (__bf16*)alloc((size_t)256 * 384 * 2);
    __bf16* bWk   = (__bf16*)alloc((size_t)256 * 384 * 2);
    __bf16* bWv   = (__bf16*)alloc((size_t)256 * 384 * 2);
    __bf16* bPe1  = (__bf16*)alloc((size_t)128 * 256 * 2);
    __bf16* bPe2  = (__bf16*)alloc((size_t)128 * 128 * 2);
    __bf16* bInW  = (__bf16*)alloc((size_t)768 * 256 * 2);
    __bf16* bOutW = (__bf16*)alloc((size_t)256 * 256 * 2);
    __bf16* bAln  = (__bf16*)alloc((size_t)256 * 512 * 2);
    __bf16* ebuf  = (__bf16*)alloc((size_t)NPTS * 256 * 2);
    __bf16* hbuf  = (__bf16*)alloc((size_t)NPTS * 128 * 2);
    __bf16* Kin   = (__bf16*)alloc((size_t)NPTS * 384 * 2);
    __bf16* Qin   = (__bf16*)alloc((size_t)NPTS * 384 * 2);
    __bf16* qb    = (__bf16*)alloc((size_t)NPTS * 256 * 2);
    __bf16* kb    = (__bf16*)alloc((size_t)NPTS * 256 * 2);
    __bf16* vb    = (__bf16*)alloc((size_t)NPTS * 256 * 2);
    __bf16* q2    = (__bf16*)alloc((size_t)NPTS * 256 * 2);
    __bf16* k2    = (__bf16*)alloc((size_t)NPTS * 256 * 2);
    __bf16* vT    = (__bf16*)alloc((size_t)NPTS * 256 * 2);
    __bf16* obuf  = (__bf16*)alloc((size_t)NPTS * 256 * 2);
    float*  oproj = (float*)alloc((size_t)NPTS * 256 * 4);
    __bf16* feat  = (__bf16*)alloc((size_t)NPTS * 512 * 2);

    auto cvt = [&](const float* s, __bf16* d, int n) {
        cvt_bf16<<<(n + 255) / 256, 256, 0, stream>>>(s, d, n);
    };
    cvt(Wq, bWq, 256 * 384);  cvt(Wk, bWk, 256 * 384);  cvt(Wv, bWv, 256 * 384);
    cvt(peW1, bPe1, 128 * 256); cvt(peW2, bPe2, 128 * 128);
    cvt(inW, bInW, 768 * 256); cvt(outW, bOutW, 256 * 256); cvt(alnW, bAln, 256 * 512);

    // --- K path: pos embed -> MLP -> Kin[:, 256:384]; x_k -> Kin[:, 0:256]
    embed_kernel<<<(NPTS * 64) / 256, 256, 0, stream>>>(icoords, ebuf, NPTS);
    gemm_wmma<EPI_RELU_BF16><<<gemm_grid(NPTS, 128), 128, 0, stream>>>(
        ebuf, bPe1, peb1, hbuf, NPTS, 128, 256, 128, 1.0f);
    gemm_wmma<EPI_BF16><<<gemm_grid(NPTS, 128), 128, 0, stream>>>(
        hbuf, bPe2, peb2, Kin + 256, NPTS, 128, 128, 384, 1.0f);
    cvt_pad<<<(NPTS * 256 + 255) / 256, 256, 0, stream>>>(inputs, Kin, NPTS, 256, 384);

    // --- Q path
    embed_kernel<<<(NPTS * 64) / 256, 256, 0, stream>>>(qcoords, ebuf, NPTS);
    gemm_wmma<EPI_RELU_BF16><<<gemm_grid(NPTS, 128), 128, 0, stream>>>(
        ebuf, bPe1, peb1, hbuf, NPTS, 128, 256, 128, 1.0f);
    gemm_wmma<EPI_BF16><<<gemm_grid(NPTS, 128), 128, 0, stream>>>(
        hbuf, bPe2, peb2, Qin + 256, NPTS, 128, 128, 384, 1.0f);
    cvt_pad<<<(NPTS * 256 + 255) / 256, 256, 0, stream>>>(Q_in, Qin, NPTS, 256, 384);

    // --- q/k/v projections (384 -> 256, no bias)
    gemm_wmma<EPI_BF16><<<gemm_grid(NPTS, 256), 128, 0, stream>>>(Qin, bWq, nullptr, qb, NPTS, 256, 384, 256, 1.0f);
    gemm_wmma<EPI_BF16><<<gemm_grid(NPTS, 256), 128, 0, stream>>>(Kin, bWk, nullptr, kb, NPTS, 256, 384, 256, 1.0f);
    gemm_wmma<EPI_BF16><<<gemm_grid(NPTS, 256), 128, 0, stream>>>(Kin, bWv, nullptr, vb, NPTS, 256, 384, 256, 1.0f);

    // --- MHA in_proj (256 -> 256, with bias); q2 pre-scaled by 1/sqrt(dh); V transposed per (b,h)
    gemm_wmma<EPI_BF16><<<gemm_grid(NPTS, 256), 128, 0, stream>>>(qb, bInW,             inb,       q2, NPTS, 256, 256, 256, 0.125f);
    gemm_wmma<EPI_BF16><<<gemm_grid(NPTS, 256), 128, 0, stream>>>(kb, bInW + 256 * 256, inb + 256, k2, NPTS, 256, 256, 256, 1.0f);
    gemm_wmma<EPI_VT_SCATTER><<<gemm_grid(NPTS, 256), 128, 0, stream>>>(vb, bInW + 512 * 256, inb + 512, vT, NPTS, 256, 256, 256, 1.0f);

    // --- SDPA: grid (B*HEADS, nq/64), 4 waves x 16 queries each
    flash_attn<<<dim3(8 * NHEADS, NPB / 64), 128, 0, stream>>>(q2, k2, vT, obuf);

    // --- out_proj (f32 out), LayerNorm, align
    gemm_wmma<EPI_F32><<<gemm_grid(NPTS, 256), 128, 0, stream>>>(obuf, bOutW, outb, oproj, NPTS, 256, 256, 256, 1.0f);
    ln_kernel<<<NPTS / 4, 128, 0, stream>>>(Q_in, oproj, lnw, lnb, feat);
    gemm_wmma<EPI_F32><<<gemm_grid(NPTS, 256), 128, 0, stream>>>(feat, bAln, nullptr, d_out, NPTS, 256, 512, 256, 1.0f);
}
